// rSCAplusplus_48120813585003
// MI455X (gfx1250) — matrix operand order repaired and checked by reading, not compiled, not run on hardware
//
#include <hip/hip_runtime.h>
#include <hip/hip_bf16.h>
#include <math.h>

typedef _Float16 v8h  __attribute__((ext_vector_type(8)));
typedef _Float16 v16h __attribute__((ext_vector_type(16)));
typedef float    v8f  __attribute__((ext_vector_type(8)));

#define B_  2
#define C_  512
#define N_  4096
#define K_  16

// ---------------------------------------------------------------------------
// Kernel 1: row norms + normalized f32 rows (both streams) + f16 copy of the
// v-stream for WMMA. Also zeroes the loss accumulator slot.
// ---------------------------------------------------------------------------
__global__ void k_normalize(const float* __restrict__ Fv, const float* __restrict__ Fi,
                            float* __restrict__ fnv, float* __restrict__ fni,
                            _Float16* __restrict__ fnh, float* __restrict__ lossslot) {
  int g = blockIdx.x * blockDim.x + threadIdx.x;   // point id in [0, B*N)
  if (g == 0) *lossslot = 0.0f;
  if (g >= B_ * N_) return;
  int b = g / N_, n = g % N_;
  const float* pv = Fv + (size_t)b * C_ * N_ + n;
  const float* pi = Fi + (size_t)b * C_ * N_ + n;
  float sv = 0.f, si = 0.f;
  for (int c = 0; c < C_; ++c) {
    float a = pv[(size_t)c * N_]; sv += a * a;
    float e = pi[(size_t)c * N_]; si += e * e;
  }
  float rv = 1.0f / (sqrtf(sv) + 1e-6f);
  float ri = 1.0f / (sqrtf(si) + 1e-6f);
  float*    ov = fnv + (size_t)g * C_;
  float*    oi = fni + (size_t)g * C_;
  _Float16* oh = fnh + (size_t)g * C_;
  for (int c = 0; c < C_; ++c) {
    float a = pv[(size_t)c * N_] * rv;
    ov[c] = a;
    oh[c] = (_Float16)a;
    oi[c] = pi[(size_t)c * N_] * ri;
  }
}

// ---------------------------------------------------------------------------
// Kernel 2: fused Gram (fn @ fn^T via v_wmma_f32_16x16x32_f16) + top-16.
// One wave owns a 16-row strip. The A strip (16x512 f16 = 16 KB) is preloaded
// once into 16 v16h register fragments (128 VGPRs). Each outer iteration
// computes 64 columns with FOUR independent WMMA accumulator chains (hides
// the XDL RAW latency), spills the 16x64 f32 tile to LDS, and all 32 lanes
// run a register-resident sorted top-16 insert (lane = row x column-half).
// Fragment layouts: 16-bit A 16x32 and B 32x16 use contiguous k = 16*hl + e in both; f32 C/D per ISA 7.12.2.
// ---------------------------------------------------------------------------
__global__ __launch_bounds__(32) void k_gram_topk(const _Float16* __restrict__ fnh,
                                                  int* __restrict__ idxout) {
  __shared__ float tile[16 * 64];     // 4 KB
  __shared__ float sV[16 * 16];
  __shared__ int   sI[16 * 16];

  const int b    = blockIdx.y;
  const int r0   = blockIdx.x * 16;
  const int lane = threadIdx.x;
  const int hl   = lane >> 4;         // 0: lanes 0-15, 1: lanes 16-31
  const int ll   = lane & 15;

  const _Float16* base = fnh + (size_t)b * N_ * C_;

  float bestv[K_];
  int   besti[K_];
#pragma unroll
  for (int q = 0; q < K_; ++q) { bestv[q] = -3.0e38f; besti[q] = 0; }

  // ---- Preload A strip: 16 k-steps of the 16x32 f16 A fragment -----------
  // lane<16:  row r0+ll, K {k0..k0+15}
  // lane>=16: row r0+ll, K {k0+16..k0+31}
  const _Float16* aptr = base + (size_t)(r0 + ll) * C_ + (hl << 4);
  v16h aF[C_ / 32];
#pragma unroll
  for (int s = 0; s < C_ / 32; ++s) {
    v8h alo = *(const v8h*)(aptr + s * 32);
    v8h ahi = *(const v8h*)(aptr + s * 32 + 8);
#pragma unroll
    for (int e = 0; e < 8; ++e) { aF[s][e] = alo[e]; aF[s][8 + e] = ahi[e]; }
  }

  for (int ct = 0; ct < N_ / 64; ++ct) {
    const int c0 = ct * 64;
    v8f acc0 = {}, acc1 = {}, acc2 = {}, acc3 = {};
    // B fragment source for tile t: column point c0+t*16+ll, 16 contiguous K
    // at k0 + hl*16 (lanes 0-15: K k0..k0+15, lanes 16-31: K k0+16..k0+31).
    const _Float16* bp0 = base + (size_t)(c0      + ll) * C_ + (hl << 4);
    const _Float16* bp1 = base + (size_t)(c0 + 16 + ll) * C_ + (hl << 4);
    const _Float16* bp2 = base + (size_t)(c0 + 32 + ll) * C_ + (hl << 4);
    const _Float16* bp3 = base + (size_t)(c0 + 48 + ll) * C_ + (hl << 4);

#pragma unroll
    for (int s = 0; s < C_ / 32; ++s) {
      const int k0 = s * 32;
      v16h b0 = *(const v16h*)(bp0 + k0);
      v16h b1 = *(const v16h*)(bp1 + k0);
      v16h b2 = *(const v16h*)(bp2 + k0);
      v16h b3 = *(const v16h*)(bp3 + k0);
      acc0 = __builtin_amdgcn_wmma_f32_16x16x32_f16(false, aF[s], false, b0, (short)0, acc0, false, false);
      acc1 = __builtin_amdgcn_wmma_f32_16x16x32_f16(false, aF[s], false, b1, (short)0, acc1, false, false);
      acc2 = __builtin_amdgcn_wmma_f32_16x16x32_f16(false, aF[s], false, b2, (short)0, acc2, false, false);
      acc3 = __builtin_amdgcn_wmma_f32_16x16x32_f16(false, aF[s], false, b3, (short)0, acc3, false, false);
    }

    // C/D layout: VGPR v, lanes 0-15 -> (M=v, N=lane); lanes 16-31 -> (M=v+8, N=lane-16)
    const int mrow = hl * 8;
#pragma unroll
    for (int v = 0; v < 8; ++v) {
      tile[(mrow + v) * 64 +      ll] = acc0[v];
      tile[(mrow + v) * 64 + 16 + ll] = acc1[v];
      tile[(mrow + v) * 64 + 32 + ll] = acc2[v];
      tile[(mrow + v) * 64 + 48 + ll] = acc3[v];
    }
    __syncthreads();

    // lane (ll, hl) scans 32 candidates of row ll, columns c0 + hl*32 + j
    const int coff = hl * 32;
#pragma unroll 8
    for (int j = 0; j < 32; ++j) {
      float v  = tile[ll * 64 + coff + j];
      int   gj = c0 + coff + j;
      if (gj != r0 + ll && v > bestv[K_ - 1]) {       // exclude self, prune
        bestv[K_ - 1] = v; besti[K_ - 1] = gj;
#pragma unroll
        for (int q = K_ - 2; q >= 0; --q) {           // one bubble pass
          if (bestv[q + 1] > bestv[q]) {
            float tv = bestv[q]; bestv[q] = bestv[q + 1]; bestv[q + 1] = tv;
            int   ti = besti[q]; besti[q] = besti[q + 1]; besti[q + 1] = ti;
          }
        }
      }
    }
    __syncthreads();
  }

  // Merge the two column-half lists per row.
  if (lane >= 16) {
#pragma unroll
    for (int q = 0; q < K_; ++q) { sV[ll * K_ + q] = bestv[q]; sI[ll * K_ + q] = besti[q]; }
  }
  __syncthreads();
  if (lane < 16) {
#pragma unroll
    for (int q = 0; q < K_; ++q) {
      float v  = sV[ll * K_ + q];
      int   gj = sI[ll * K_ + q];
      if (v > bestv[K_ - 1]) {
        bestv[K_ - 1] = v; besti[K_ - 1] = gj;
#pragma unroll
        for (int p = K_ - 2; p >= 0; --p) {
          if (bestv[p + 1] > bestv[p]) {
            float tv = bestv[p]; bestv[p] = bestv[p + 1]; bestv[p + 1] = tv;
            int   ti = besti[p]; besti[p] = besti[p + 1]; besti[p + 1] = ti;
          }
        }
      }
    }
    int* o = idxout + ((size_t)b * N_ + r0 + ll) * K_;
#pragma unroll
    for (int q = 0; q < K_; ++q) o[q] = besti[q];     // descending similarity
  }
}

// ---------------------------------------------------------------------------
// Kernel 3: exact f32 coherence for the selected neighbors, both streams.
// ---------------------------------------------------------------------------
__global__ void k_coherence(const float* __restrict__ fnv, const float* __restrict__ fni,
                            const int* __restrict__ idx, const float* __restrict__ temp,
                            float* __restrict__ cv, float* __restrict__ ci) {
  int g = blockIdx.x * blockDim.x + threadIdx.x;     // (b, n, k) flat
  if (g >= B_ * N_ * K_) return;
  int b = g / (N_ * K_);
  int n = (g / K_) % N_;
  int j = idx[g];
  const float4* pv = (const float4*)(fnv + ((size_t)b * N_ + n) * C_);
  const float4* qv = (const float4*)(fnv + ((size_t)b * N_ + j) * C_);
  const float4* pi = (const float4*)(fni + ((size_t)b * N_ + n) * C_);
  const float4* qi = (const float4*)(fni + ((size_t)b * N_ + j) * C_);
  float dv = 0.f, di = 0.f;
  for (int c = 0; c < C_ / 4; ++c) {
    float4 a = pv[c], bb = qv[c];
    dv += a.x * bb.x + a.y * bb.y + a.z * bb.z + a.w * bb.w;
    float4 e = pi[c], f = qi[c];
    di += e.x * f.x + e.y * f.y + e.z * f.z + e.w * f.w;
  }
  float it = 1.0f / temp[0];
  cv[g] = dv * it;
  ci[g] = di * it;
}

// ---------------------------------------------------------------------------
// Kernel 4: MLP gate (16->32->1, relu, sigmoid) per point + alignment loss.
// ---------------------------------------------------------------------------
__global__ void k_gate_loss(const float* __restrict__ cv, const float* __restrict__ ci,
                            const float* __restrict__ W1, const float* __restrict__ b1,
                            const float* __restrict__ W2, const float* __restrict__ b2,
                            float* __restrict__ gv, float* __restrict__ gi,
                            float* __restrict__ loss) {
  __shared__ float red[256];
  int g = blockIdx.x * blockDim.x + threadIdx.x;     // point id
  float part = 0.f;
  if (g < B_ * N_) {
    float c_v[K_], c_i[K_];
#pragma unroll
    for (int k = 0; k < K_; ++k) { c_v[k] = cv[(size_t)g * K_ + k]; c_i[k] = ci[(size_t)g * K_ + k]; }

    float acc_v = b2[0], acc_i = b2[0];
    for (int j = 0; j < 32; ++j) {
      float hv = b1[j], hi = b1[j];
#pragma unroll
      for (int k = 0; k < K_; ++k) { float w = W1[k * 32 + j]; hv += c_v[k] * w; hi += c_i[k] * w; }
      hv = fmaxf(hv, 0.f); hi = fmaxf(hi, 0.f);
      float w2 = W2[j];
      acc_v += hv * w2; acc_i += hi * w2;
    }
    gv[g] = 1.0f / (1.0f + expf(-acc_v));
    gi[g] = 1.0f / (1.0f + expf(-acc_i));

    float sv = 0.f, si = 0.f;
#pragma unroll
    for (int k = 0; k < K_; ++k) { sv += c_v[k] * c_v[k]; si += c_i[k] * c_i[k]; }
    float rv = 1.0f / fmaxf(sqrtf(sv), 1e-12f);
    float ri = 1.0f / fmaxf(sqrtf(si), 1e-12f);
#pragma unroll
    for (int k = 0; k < K_; ++k) { float d = c_v[k] * rv - c_i[k] * ri; part += d * d; }
  }
  red[threadIdx.x] = part;
  __syncthreads();
  for (int s = 128; s > 0; s >>= 1) {
    if (threadIdx.x < s) red[threadIdx.x] += red[threadIdx.x + s];
    __syncthreads();
  }
  if (threadIdx.x == 0) atomicAdd(loss, red[0] * (100.0f / (float)(B_ * N_ * K_)));
}

// ---------------------------------------------------------------------------
// Kernel 5: gate broadcast scaling into [B,C,H,W] outputs (coalesced over n).
// ---------------------------------------------------------------------------
__global__ void k_scale(const float* __restrict__ Fv, const float* __restrict__ Fi,
                        const float* __restrict__ gv, const float* __restrict__ gi,
                        float* __restrict__ out) {
  size_t g = (size_t)blockIdx.x * blockDim.x + threadIdx.x;
  const size_t BCN = (size_t)B_ * C_ * N_;
  if (g >= 2 * BCN) return;
  if (g < BCN) {
    size_t b = g / ((size_t)C_ * N_);
    size_t n = g % N_;
    out[g] = Fv[g] * gv[b * N_ + n];
  } else {
    size_t r = g - BCN;
    size_t b = r / ((size_t)C_ * N_);
    size_t n = r % N_;
    out[g] = Fi[r] * gi[b * N_ + n];
  }
}

extern "C" void kernel_launch(void* const* d_in, const int* in_sizes, int n_in,
                              void* d_out, int out_size, void* d_ws, size_t ws_size,
                              hipStream_t stream) {
  const float* Fv   = (const float*)d_in[0];
  const float* Fi   = (const float*)d_in[1];
  const float* W1   = (const float*)d_in[2];
  const float* b1   = (const float*)d_in[3];
  const float* W2   = (const float*)d_in[4];
  const float* b2   = (const float*)d_in[5];
  const float* temp = (const float*)d_in[6];
  float* out = (float*)d_out;

  const size_t BNC = (size_t)B_ * N_ * C_;          // 4,194,304
  const size_t BNK = (size_t)B_ * N_ * K_;          // 131,072
  const size_t BN  = (size_t)B_ * N_;               // 8,192
  char* ws = (char*)d_ws;

  float*    fnv = (float*)(ws);                     // 16 MB  normalized v (f32)
  float*    fni = (float*)(ws + BNC * 4);           // 16 MB  normalized i (f32)
  _Float16* fnh = (_Float16*)(ws + BNC * 8);        //  8 MB  normalized v (f16)
  int*      idx = (int*)(ws + BNC * 10);            // 512 KB KNN indices
  float*    cv  = (float*)(ws + BNC * 10 + BNK * 4);
  float*    ci  = (float*)(ws + BNC * 10 + BNK * 8);
  float*    gv  = (float*)(ws + BNC * 10 + BNK * 12);
  float*    gi  = (float*)(ws + BNC * 10 + BNK * 12 + BN * 4);

  float* loss = out + 2 * (size_t)B_ * C_ * N_;     // scalar slot at the end

  k_normalize<<<(B_ * N_ + 255) / 256, 256, 0, stream>>>(Fv, Fi, fnv, fni, fnh, loss);

  dim3 g2(N_ / 16, B_);
  k_gram_topk<<<g2, 32, 0, stream>>>(fnh, idx);

  k_coherence<<<(int)((BNK + 127) / 128), 128, 0, stream>>>(fnv, fni, idx, temp, cv, ci);

  k_gate_loss<<<(B_ * N_ + 255) / 256, 256, 0, stream>>>(cv, ci, W1, b1, W2, b2, gv, gi, loss);

  size_t tot = 2 * (size_t)B_ * C_ * N_;
  k_scale<<<(int)((tot + 255) / 256), 256, 0, stream>>>(Fv, Fi, gv, gi, out);
}
